// PolicyNetwork_63857573757066
// MI455X (gfx1250) — matrix operand order repaired and checked by reading, not compiled, not run on hardware
//
#include <hip/hip_runtime.h>
#include <hip/hip_bf16.h>

// CDNA5 (gfx1250) fused policy-network pipeline, round 3.
// - bf16 WMMA (v_wmma_f32_16x16x32_bf16), fp32 accumulate, M=32 register blocking
// - 16B-aligned LDS fragment rows (stride 40) -> ds_load_b128 fragment loads
// - B fragments software-pipelined one step ahead of the WMMA pair
// - LayerNorm from accumulator registers (shuffle butterflies + tiny LDS)

typedef __attribute__((ext_vector_type(16))) __bf16 v16bf;
typedef __attribute__((ext_vector_type(8)))  float  v8f;

#define BT_STRIDE 40   // 80B row pitch: 16B-aligned fragments, <=2-way bank conflicts
#define TPB 256
#define MT  32         // rows per workgroup
#define MF  (MT / 16)  // A fragments (M blocks)

__device__ __forceinline__ unsigned short f2bf(float f) {
  unsigned int u = __float_as_uint(f);
  u += 0x7FFFu + ((u >> 16) & 1u);     // round-to-nearest-even
  return (unsigned short)(u >> 16);
}

union FragU { unsigned int u[8]; uint4 q[2]; v16bf v; };

// OUT[row0+m][n] = act( X[MT x KDIM] @ W[KDIM x NDIM] + bias ), optional LN(+residual).
template<int KDIM, int NDIM, bool RELU, bool LN, bool RES>
__global__ __launch_bounds__(TPB)
void gemm_fused(const float* __restrict__ X, const float* __restrict__ W,
                const float* __restrict__ bias, const float* __restrict__ gamma,
                const float* __restrict__ beta, const float* __restrict__ RESIN,
                float* __restrict__ OUT)
{
  extern __shared__ unsigned char smem_raw[];
  unsigned short* bt = (unsigned short*)smem_raw;                      // [NDIM][BT_STRIDE] bf16 W^T tile
  unsigned short* sA = (unsigned short*)(smem_raw + (size_t)NDIM * BT_STRIDE * 2); // [MT][32] bf16
  float* psum  = (float*)(smem_raw + (size_t)NDIM * BT_STRIDE * 2 + MT * 32 * 2);  // [MT][8]
  float* psum2 = psum  + MT * 8;                                       // [MT][8]
  float* mrow  = psum2 + MT * 8;                                       // [MT]
  float* rrow  = mrow  + MT;                                           // [MT]

  const int tid  = threadIdx.x;
  const int lane = tid & 31;
  const int wave = tid >> 5;
  const int row0 = blockIdx.x * MT;

  constexpr int FPW = NDIM / (8 * 16);   // 16-col fragments per wave
  const int nwave0 = wave * FPW * 16;

  v8f acc[MF][FPW];
  const v8f vzero = {0.f,0.f,0.f,0.f,0.f,0.f,0.f,0.f};
  #pragma unroll
  for (int mf = 0; mf < MF; ++mf)
    #pragma unroll
    for (int f = 0; f < FPW; ++f) acc[mf][f] = vzero;

  const int m    = lane & 15;
  const int half = (lane < 16) ? 0 : 1;
  const int kb0  = half ? 8 : 0;     // A-frag K origin of first 16-K group
  const int khB  = half ? 16 : 0;    // unused; B frags also load at kb0

  constexpr int KSTEPS = KDIM / 32;
  for (int kt = 0; kt < KSTEPS; ++kt) {
    __syncthreads();
    // Stage A tile [MT][32] fp32 -> bf16 (coalesced along K)
    for (int i = tid; i < MT * 32; i += TPB) {
      int r = i >> 5, c = i & 31;
      sA[i] = f2bf(X[(size_t)(row0 + r) * KDIM + (kt * 32 + c)]);
    }
    // Stage W^T tile: bt[n][k] = bf16(W[kt*32+k][n]); global reads coalesced along n
    for (int i = tid; i < 32 * NDIM; i += TPB) {
      int k = i / NDIM, n = i % NDIM;
      bt[n * BT_STRIDE + k] = f2bf(W[(size_t)(kt * 32 + k) * NDIM + n]);
      if (kt + 1 < KSTEPS && k == 0)   // prefetch next K-tile (global_prefetch_b8)
        __builtin_prefetch(&W[(size_t)((kt + 1) * 32) * NDIM + n], 0, 0);
    }
    __syncthreads();

    // A fragments (one per 16-row block), 2x ds_load_b128 each
    FragU af[MF];
    #pragma unroll
    for (int mf = 0; mf < MF; ++mf) {
      af[mf].q[0] = *(const uint4*)&sA[(mf * 16 + m) * 32 + kb0];
      af[mf].q[1] = *(const uint4*)&sA[(mf * 16 + m) * 32 + kb0 + 16];
    }
    // B fragments: software-pipelined one step ahead; each feeds MF WMMAs
    FragU bcur, bnxt;
    {
      const int n0 = nwave0 + m;
      bcur.q[0] = *(const uint4*)&bt[n0 * BT_STRIDE + kb0];
      bcur.q[1] = *(const uint4*)&bt[n0 * BT_STRIDE + kb0 + 16];
    }
    #pragma unroll
    for (int f = 0; f < FPW; ++f) {
      if (f + 1 < FPW) {
        const int n1 = nwave0 + (f + 1) * 16 + m;
        bnxt.q[0] = *(const uint4*)&bt[n1 * BT_STRIDE + kb0];
        bnxt.q[1] = *(const uint4*)&bt[n1 * BT_STRIDE + kb0 + 16];
      }
      #pragma unroll
      for (int mf = 0; mf < MF; ++mf)
        acc[mf][f] = __builtin_amdgcn_wmma_f32_16x16x32_bf16(
                       false, af[mf].v, false, bcur.v, (short)0, acc[mf][f], false, false);
      bcur = bnxt;
    }
  }

  // bias (+ReLU) applied in-registers; C layout: VGPR v -> M = v (+8 for upper half)
  #pragma unroll
  for (int mf = 0; mf < MF; ++mf)
    #pragma unroll
    for (int f = 0; f < FPW; ++f) {
      const int n = nwave0 + f * 16 + m;
      const float bn = bias[n];
      #pragma unroll
      for (int v = 0; v < 8; ++v) {
        float val = acc[mf][f][v] + bn;
        if (RELU) val = fmaxf(val, 0.f);
        acc[mf][f][v] = val;
      }
    }

  if constexpr (LN) {
    // Per-row partial sums over this wave's columns; xor masks < 16 reduce
    // within the 16-lane half that shares the row.
    #pragma unroll
    for (int mf = 0; mf < MF; ++mf)
      #pragma unroll
      for (int v = 0; v < 8; ++v) {
        float s = 0.f, s2 = 0.f;
        #pragma unroll
        for (int f = 0; f < FPW; ++f) { float x = acc[mf][f][v]; s += x; s2 += x * x; }
        #pragma unroll
        for (int off = 8; off >= 1; off >>= 1) {
          s  += __shfl_xor(s,  off, 32);
          s2 += __shfl_xor(s2, off, 32);
        }
        if (m == 0) {                       // lanes 0 and 16 write their row
          const int row = mf * 16 + v + half * 8;
          psum [row * 8 + wave] = s;
          psum2[row * 8 + wave] = s2;
        }
      }
    __syncthreads();
    if (tid < MT) {                         // one thread per row: cross-wave reduce
      float s = 0.f, s2 = 0.f;
      #pragma unroll
      for (int w = 0; w < 8; ++w) { s += psum[tid * 8 + w]; s2 += psum2[tid * 8 + w]; }
      const float mean = s * (1.f / NDIM);
      const float var  = s2 * (1.f / NDIM) - mean * mean;
      mrow[tid] = mean;
      rrow[tid] = rsqrtf(var + 1e-5f);
    }
    __syncthreads();
    // Normalize from registers and write out (+ residual)
    #pragma unroll
    for (int mf = 0; mf < MF; ++mf)
      #pragma unroll
      for (int f = 0; f < FPW; ++f) {
        const int n = nwave0 + f * 16 + m;
        const float g = gamma[n], be = beta[n];
        #pragma unroll
        for (int v = 0; v < 8; ++v) {
          const int row = mf * 16 + v + half * 8;
          float o = (acc[mf][f][v] - mrow[row]) * rrow[row] * g + be;
          if (RES) o += RESIN[(size_t)(row0 + row) * NDIM + n];
          OUT[(size_t)(row0 + row) * NDIM + n] = o;
        }
      }
  } else {
    #pragma unroll
    for (int mf = 0; mf < MF; ++mf)
      #pragma unroll
      for (int f = 0; f < FPW; ++f) {
        const int n = nwave0 + f * 16 + m;
        #pragma unroll
        for (int v = 0; v < 8; ++v) {
          const int row = mf * 16 + v + half * 8;
          OUT[(size_t)(row0 + row) * NDIM + n] = acc[mf][f][v];
        }
      }
  }
}

// logits[b][a] = dot(z[b], AE[b][a]) ; mask(sum idx == 0) -> -inf
__global__ __launch_bounds__(TPB)
void logits_kernel(const float* __restrict__ Z, const float* __restrict__ AE,
                   const int* __restrict__ IDX, float* __restrict__ OUT)
{
  __shared__ float sz[256];
  const int b = blockIdx.x;
  const int tid = threadIdx.x, lane = tid & 31, wave = tid >> 5;
  sz[tid] = Z[(size_t)b * 256 + tid];
  __syncthreads();
  for (int it = 0; it < 32; ++it) {
    const int a = wave * 32 + it;
    const float* ae = AE + ((size_t)b * 256 + a) * 256;
    float p = 0.f;
    #pragma unroll
    for (int i = 0; i < 8; ++i) {
      const int e = lane + 32 * i;          // coalesced across the wave
      p += ae[e] * sz[e];
    }
    #pragma unroll
    for (int off = 16; off > 0; off >>= 1) p += __shfl_xor(p, off, 32);
    if (lane == 0) {
      const int* ix = IDX + ((size_t)b * 256 + a) * 4;
      const int ssum = ix[0] + ix[1] + ix[2] + ix[3];
      OUT[(size_t)b * 256 + a] = (ssum == 0) ? -__builtin_huge_valf() : p;
    }
  }
}

extern "C" void kernel_launch(void* const* d_in, const int* in_sizes, int n_in,
                              void* d_out, int out_size, void* d_ws, size_t ws_size,
                              hipStream_t stream) {
  (void)in_sizes; (void)n_in; (void)out_size; (void)ws_size;
  const float* obs = (const float*)d_in[0];
  const float* AE  = (const float*)d_in[1];
  const int*   IDX = (const int*)d_in[2];
  const float* w0  = (const float*)d_in[3];
  const float* b0  = (const float*)d_in[4];
  const float* g0  = (const float*)d_in[5];
  const float* be0 = (const float*)d_in[6];
  const float* Wb  = (const float*)d_in[7];
  const float* bb  = (const float*)d_in[8];
  const float* gb  = (const float*)d_in[9];
  const float* beb = (const float*)d_in[10];
  const float* w1  = (const float*)d_in[11];
  const float* b1  = (const float*)d_in[12];
  const float* w2  = (const float*)d_in[13];
  const float* b2  = (const float*)d_in[14];

  const int B = 16384, H = 1024, E = 256;
  float* x0 = (float*)d_ws;
  float* x1 = x0 + (size_t)B * H;
  float* z  = x1 + (size_t)B * H;

  dim3 grid(B / MT), blk(TPB);
  auto smem = [](int ndim) {
    return (size_t)ndim * BT_STRIDE * 2      // bt
         + MT * 32 * 2                       // sA
         + MT * 8 * 4 * 2                    // psum, psum2
         + MT * 4 * 2;                       // mrow, rrow
  };
  const size_t smemH = smem(H);   // ~84.5 KB (< 320 KB WGP LDS)
  const size_t smemE = smem(E);   // ~24.5 KB

  // obs_transform: Linear(E,H) + ReLU + LN
  gemm_fused<256, 1024, true, true, false><<<grid, blk, smemH, stream>>>(
      obs, w0, b0, g0, be0, nullptr, x0);

  // 8 residual blocks, ping-pong x0/x1
  const float* xin = x0; float* xout = x1;
  for (int i = 0; i < 8; ++i) {
    gemm_fused<1024, 1024, true, true, true><<<grid, blk, smemH, stream>>>(
        xin, Wb + (size_t)i * H * H, bb + (size_t)i * H, gb + (size_t)i * H,
        beb + (size_t)i * H, xin, xout);
    float* t = (float*)xin; xin = xout; xout = t;
  }
  // after 8 swaps: xin == x0, xout == x1
  gemm_fused<1024, 1024, true, false, false><<<grid, blk, smemH, stream>>>(
      xin, w1, b1, nullptr, nullptr, nullptr, xout);
  gemm_fused<1024, 256, false, false, false><<<grid, blk, smemE, stream>>>(
      xout, w2, b2, nullptr, nullptr, nullptr, z);

  logits_kernel<<<dim3(B), blk, 0, stream>>>(z, AE, IDX, (float*)d_out);
}
